// Net_22256520528693
// MI455X (gfx1250) — matrix operand order repaired and checked by reading, not compiled, not run on hardware
//
#include <hip/hip_runtime.h>

// ---------------------------------------------------------------------------
// Problem constants (reference: BS=8, N=200, UNITS=128, DEPTH=3)
// ---------------------------------------------------------------------------
#define BSZ    8
#define NN     200
#define UU     128
#define E0     39800            // NN*(NN-1)
#define ET     318400           // BSZ*E0  (multiple of 16: 16*19900)
#define NODES  1600             // BSZ*NN
#define DEPTH  3
#define NTILE  (ET/16)          // 19900 row tiles for the edge GEMM
#define TILES_PER_WG 8
#define EGEMM_WGS ((NTILE + TILES_PER_WG - 1) / TILES_PER_WG)   // 2488

typedef __attribute__((ext_vector_type(16))) __bf16 v16bf;
typedef __attribute__((ext_vector_type(8)))  float  v8f;

union BF16Frag { v16bf v; uint4 q[2]; };

__device__ __forceinline__ unsigned int f32_to_bf16(float f) {
    unsigned int u = __float_as_uint(f);
    unsigned int r = (u + 0x7FFFu + ((u >> 16) & 1u)) >> 16;   // RNE
    return r & 0xFFFFu;
}

__device__ __forceinline__ float relu_f(float x) { return fmaxf(x, 0.0f); }

// ---------------------------------------------------------------------------
// Setup kernels
// ---------------------------------------------------------------------------
struct W15 { const float* p[15]; };   // [layer][v1,v2,v3,v4,e] f32 128x128

__global__ void k_prep_weights(W15 ws, unsigned short* __restrict__ out) {
    int idx = blockIdx.x * 256 + threadIdx.x;        // 15 * 16384 total
    int m = idx >> 14, j = idx & 16383;
    out[idx] = (unsigned short)f32_to_bf16(ws.p[m][j]);
}

__global__ void k_build_edges(int* __restrict__ src, int* __restrict__ dst) {
    int e = blockIdx.x * 256 + threadIdx.x;
    if (e >= ET) return;
    int b = e / E0;
    int q = e - b * E0;
    int u = q / 199;
    int r = q - u * 199;
    int v = r + (r >= u);
    src[e] = b * NN + u;
    dst[e] = b * NN + v;
}

// xv = relu(x @ W0[2,128] + b0)
__global__ void k_node_embed(const float* __restrict__ x,
                             const float* __restrict__ w0,
                             const float* __restrict__ b0,
                             float* __restrict__ xv) {
    int idx = blockIdx.x * 256 + threadIdx.x;        // NODES*UU = 204800
    int r = idx >> 7, c = idx & 127;
    float v = x[r * 2] * w0[c] + x[r * 2 + 1] * w0[UU + c] + b0[c];
    xv[idx] = relu_f(v);
}

// w = relu(adj[b,u,v] * we0[c] + be0[c])
__global__ void k_edge_embed(const float* __restrict__ adj,
                             const float* __restrict__ we0,
                             const float* __restrict__ be0,
                             float* __restrict__ w) {
    long long idx = (long long)blockIdx.x * 256 + threadIdx.x;  // ET*UU
    int e = (int)(idx >> 7), c = (int)(idx & 127);
    int b = e / E0;
    int q = e - b * E0;
    int u = q / 199;
    int r = q - u * 199;
    int v = r + (r >= u);
    float a = adj[(size_t)b * NN * NN + u * NN + v];
    w[idx] = relu_f(a * we0[c] + be0[c]);
}

__global__ void k_zero(float* __restrict__ p, int n) {
    int i = blockIdx.x * 256 + threadIdx.x;
    if (i < n) p[i] = 0.0f;
}

// ---------------------------------------------------------------------------
// Node GEMM: out[wsel] = relu-free  (x0 @ W[wsel] + bias[wsel]),  M=1600
// grid = (100, 4), block = 256 (8 waves; wave w -> columns 16w..16w+15)
// ---------------------------------------------------------------------------
struct NodeGemmArgs {
    const float* A;                       // xv [NODES][UU] f32
    const unsigned short* W[4];           // bf16 [UU][UU]
    const float* bias[4];
    float* out[4];
};

__global__ __launch_bounds__(256) void k_node_gemm(NodeGemmArgs ar) {
    __shared__ __align__(16) unsigned short ldsA[16][136];
    const int tid  = threadIdx.x;
    const int lane = tid & 31, wave = tid >> 5;
    const int n0   = wave * 16, nloc = lane & 15, col = n0 + nloc;
    const int wsel = blockIdx.y, tile = blockIdx.x;

    const unsigned short* W = ar.W[wsel];
    BF16Frag bfrag[4];
#pragma unroll
    for (int kk = 0; kk < 4; ++kk) {
        unsigned int pk[8];
#pragma unroll
        for (int p = 0; p < 8; ++p) {
            int k = kk * 32 + ((lane >> 4) << 3) + ((p & 3) << 1) + ((p >> 2) << 4);
            pk[p] = (unsigned int)W[(size_t)k * UU + col] | ((unsigned int)W[(size_t)(k + 1) * UU + col] << 16);
        }
        bfrag[kk].q[0] = make_uint4(pk[0], pk[1], pk[2], pk[3]);
        bfrag[kk].q[1] = make_uint4(pk[4], pk[5], pk[6], pk[7]);
    }
    const float bval = ar.bias[wsel][col];

    {   // stage A tile (rows tile*16 .. +15), f32 -> bf16
        int row = tid >> 4, c8 = (tid & 15) * 8;
        const float4* gp = (const float4*)(ar.A + (size_t)(tile * 16 + row) * UU + c8);
        float4 f0 = gp[0], f1 = gp[1];
        uint4 o;
        o.x = f32_to_bf16(f0.x) | (f32_to_bf16(f0.y) << 16);
        o.y = f32_to_bf16(f0.z) | (f32_to_bf16(f0.w) << 16);
        o.z = f32_to_bf16(f1.x) | (f32_to_bf16(f1.y) << 16);
        o.w = f32_to_bf16(f1.z) | (f32_to_bf16(f1.w) << 16);
        *(uint4*)&ldsA[row][c8] = o;
    }
    __syncthreads();

    const int arow  = nloc;
    const int khalf = (lane >> 4) << 3;          // 0 or 8
    v8f acc;
#pragma unroll
    for (int i = 0; i < 8; ++i) acc[i] = bval;
#pragma unroll
    for (int kk = 0; kk < 4; ++kk) {
        BF16Frag af;
        af.q[0] = *(const uint4*)&ldsA[arow][kk * 32 + khalf];
        af.q[1] = *(const uint4*)&ldsA[arow][kk * 32 + khalf + 16];
        acc = __builtin_amdgcn_wmma_f32_16x16x32_bf16(
                  false, af.v, false, bfrag[kk].v, (short)0, acc, false, false);
    }
    const int mbase = tile * 16 + ((lane >> 4) << 3);
    float* out = ar.out[wsel];
#pragma unroll
    for (int i = 0; i < 8; ++i)
        out[(size_t)(mbase + i) * UU + col] = acc[i];
}

// ---------------------------------------------------------------------------
// Edge GEMM fused epilogue:
//   t = w0 @ We + be + x3[src] + x4[dst];  accumulate Σt, Σt² per column
// grid = EGEMM_WGS, block = 256
// ---------------------------------------------------------------------------
__global__ __launch_bounds__(256) void k_edge_gemm(
        const float* __restrict__ w0,
        const unsigned short* __restrict__ We,
        const float* __restrict__ be,
        const float* __restrict__ x3,
        const float* __restrict__ x4,
        const int* __restrict__ src,
        const int* __restrict__ dst,
        float* __restrict__ t,
        float* __restrict__ esum,
        float* __restrict__ esumsq) {
    __shared__ __align__(16) unsigned short ldsA[16][136];
    __shared__ float sSum[UU], sSq[UU];
    const int tid  = threadIdx.x;
    const int lane = tid & 31, wave = tid >> 5;
    const int n0   = wave * 16, nloc = lane & 15, col = n0 + nloc;

    if (tid < UU) { sSum[tid] = 0.0f; sSq[tid] = 0.0f; }

    BF16Frag bfrag[4];
#pragma unroll
    for (int kk = 0; kk < 4; ++kk) {
        unsigned int pk[8];
#pragma unroll
        for (int p = 0; p < 8; ++p) {
            int k = kk * 32 + ((lane >> 4) << 3) + ((p & 3) << 1) + ((p >> 2) << 4);
            pk[p] = (unsigned int)We[(size_t)k * UU + col] | ((unsigned int)We[(size_t)(k + 1) * UU + col] << 16);
        }
        bfrag[kk].q[0] = make_uint4(pk[0], pk[1], pk[2], pk[3]);
        bfrag[kk].q[1] = make_uint4(pk[4], pk[5], pk[6], pk[7]);
    }
    const float bval = be[col];

    for (int it = 0; it < TILES_PER_WG; ++it) {
        const int tile = blockIdx.x * TILES_PER_WG + it;
        __syncthreads();
        if (tile < NTILE) {   // stage A tile, f32 -> bf16
            int row = tid >> 4, c8 = (tid & 15) * 8;
            const float4* gp =
                (const float4*)(w0 + (size_t)(tile * 16 + row) * UU + c8);
            float4 f0 = gp[0], f1 = gp[1];
            uint4 o;
            o.x = f32_to_bf16(f0.x) | (f32_to_bf16(f0.y) << 16);
            o.y = f32_to_bf16(f0.z) | (f32_to_bf16(f0.w) << 16);
            o.z = f32_to_bf16(f1.x) | (f32_to_bf16(f1.y) << 16);
            o.w = f32_to_bf16(f1.z) | (f32_to_bf16(f1.w) << 16);
            *(uint4*)&ldsA[row][c8] = o;
        }
        __syncthreads();
        if (tile >= NTILE) continue;

        const int arow  = nloc;
        const int khalf = (lane >> 4) << 3;
        v8f acc;
#pragma unroll
        for (int i = 0; i < 8; ++i) acc[i] = bval;
#pragma unroll
        for (int kk = 0; kk < 4; ++kk) {
            BF16Frag af;
            af.q[0] = *(const uint4*)&ldsA[arow][kk * 32 + khalf];
            af.q[1] = *(const uint4*)&ldsA[arow][kk * 32 + khalf + 16];
            acc = __builtin_amdgcn_wmma_f32_16x16x32_bf16(
                      false, af.v, false, bfrag[kk].v, (short)0, acc, false, false);
        }

        const int mbase = tile * 16 + ((lane >> 4) << 3);
        float ls = 0.0f, lq = 0.0f;
#pragma unroll
        for (int i = 0; i < 8; ++i) {
            int mg = mbase + i;
            float val = acc[i] + x3[(size_t)src[mg] * UU + col]
                               + x4[(size_t)dst[mg] * UU + col];
            t[(size_t)mg * UU + col] = val;
            ls += val;
            lq += val * val;
        }
        atomicAdd(&sSum[col], ls);
        atomicAdd(&sSq[col], lq);
    }
    __syncthreads();
    if (tid < UU) {
        atomicAdd(&esum[tid], sSum[tid]);
        atomicAdd(&esumsq[tid], sSq[tid]);
    }
}

// ---------------------------------------------------------------------------
// agg[b,u,c] = max over v!=u of sigmoid(w0[b,u,v,c]) * x2[b,v,c]
// Edges of (b,u) are contiguous (u-major) -> no atomics. grid=NODES, blk=128
// ---------------------------------------------------------------------------
__global__ __launch_bounds__(128) void k_agg(const float* __restrict__ w0,
                                             const float* __restrict__ x2,
                                             float* __restrict__ agg) {
    const int c  = threadIdx.x;
    const int un = blockIdx.x;
    const int b  = un / NN, u = un - b * NN;
    const float* wb = w0 + ((size_t)(b * E0 + u * 199)) * UU + c;
    const float* xb = x2 + (size_t)b * NN * UU + c;
    float m = -3.402823e38f;
    for (int r = 0; r < 199; ++r) {
        int v = r + (r >= u);
        float s  = 1.0f / (1.0f + expf(-wb[(size_t)r * UU]));
        float vl = s * xb[(size_t)v * UU];
        m = fmaxf(m, vl);
    }
    agg[(size_t)un * UU + c] = m;
}

// BN stats over nodes: column c of (x1+agg). grid=128, block=256
__global__ __launch_bounds__(256) void k_node_stats(const float* __restrict__ x1,
                                                    const float* __restrict__ agg,
                                                    float* __restrict__ nmu,
                                                    float* __restrict__ nrsig) {
    __shared__ float s1[256], s2[256];
    const int c = blockIdx.x, tid = threadIdx.x;
    float s = 0.0f, q = 0.0f;
    for (int r = tid; r < NODES; r += 256) {
        float y = x1[(size_t)r * UU + c] + agg[(size_t)r * UU + c];
        s += y; q += y * y;
    }
    s1[tid] = s; s2[tid] = q;
    __syncthreads();
    for (int o = 128; o; o >>= 1) {
        if (tid < o) { s1[tid] += s1[tid + o]; s2[tid] += s2[tid + o]; }
        __syncthreads();
    }
    if (tid == 0) {
        float mu  = s1[0] / (float)NODES;
        float var = s2[0] / (float)NODES - mu * mu;
        nmu[c]   = mu;
        nrsig[c] = rsqrtf(var + 1e-5f);
    }
}

__global__ void k_edge_stats_final(const float* __restrict__ esum,
                                   const float* __restrict__ esumsq,
                                   float* __restrict__ emu,
                                   float* __restrict__ ersig) {
    int c = threadIdx.x;     // 128 threads
    float mu  = esum[c] / (float)ET;
    float var = esumsq[c] / (float)ET - mu * mu;
    emu[c]   = mu;
    ersig[c] = rsqrtf(var + 1e-5f);
}

// xv += relu( g*(x1+agg-mu)*rsig + b )
__global__ void k_node_update(float* __restrict__ xv,
                              const float* __restrict__ x1,
                              const float* __restrict__ agg,
                              const float* __restrict__ nmu,
                              const float* __restrict__ nrsig,
                              const float* __restrict__ g,
                              const float* __restrict__ bt) {
    int idx = blockIdx.x * 256 + threadIdx.x;      // NODES*UU
    int c = idx & 127;
    float y = x1[idx] + agg[idx];
    float h = g[c] * (y - nmu[c]) * nrsig[c] + bt[c];
    xv[idx] = xv[idx] + relu_f(h);
}

// w += relu( g*(t-mu)*rsig + b )
__global__ void k_edge_update(float* __restrict__ w,
                              const float* __restrict__ t,
                              const float* __restrict__ emu,
                              const float* __restrict__ ersig,
                              const float* __restrict__ g,
                              const float* __restrict__ bt) {
    long long idx = (long long)blockIdx.x * 256 + threadIdx.x;   // ET*UU
    int c = (int)(idx & 127);
    float h = g[c] * (t[idx] - emu[c]) * ersig[c] + bt[c];
    w[idx] = w[idx] + relu_f(h);
}

// out[b,u,v] = w[e,:] . we1 + be1   (one wave32 per edge, float4 + shfl_xor)
__global__ __launch_bounds__(256) void k_out(const float* __restrict__ w,
                                             const float* __restrict__ we1,
                                             const float* __restrict__ be1,
                                             float* __restrict__ out) {
    const int lane = threadIdx.x & 31;
    const int wv   = threadIdx.x >> 5;
    const long long e = (long long)blockIdx.x * 8 + wv;   // grid = ET/8
    const float4* wp = (const float4*)(w + (size_t)e * UU);
    const float4* vp = (const float4*)we1;
    float4 a = wp[lane], b4 = vp[lane];
    float s = a.x * b4.x + a.y * b4.y + a.z * b4.z + a.w * b4.w;
    for (int o = 16; o; o >>= 1) s += __shfl_xor(s, o, 32);
    if (lane == 0) {
        int ei = (int)e;
        int b = ei / E0;
        int q = ei - b * E0;
        int u = q / 199;
        int r = q - u * 199;
        int v = r + (r >= u);
        out[(size_t)b * NN * NN + u * NN + v] = s + be1[0];
    }
}

__global__ void k_out_diag(float* __restrict__ out) {
    int idx = blockIdx.x * 256 + threadIdx.x;
    if (idx >= NODES) return;
    int b = idx / NN, u = idx - b * NN;
    out[(size_t)b * NN * NN + u * NN + u] = 0.0f;
}

// ---------------------------------------------------------------------------
// Launch
// ---------------------------------------------------------------------------
extern "C" void kernel_launch(void* const* d_in, const int* in_sizes, int n_in,
                              void* d_out, int out_size, void* d_ws, size_t ws_size,
                              hipStream_t stream) {
    (void)in_sizes; (void)n_in; (void)out_size; (void)ws_size;

    // --- input map (setup_inputs() insertion order) ---
    const float* x    = (const float*)d_in[0];
    const float* adj  = (const float*)d_in[1];
    const float* vl0w = (const float*)d_in[2];
    const float* vl0b = (const float*)d_in[3];
    const float* el0w = (const float*)d_in[4];
    const float* el0b = (const float*)d_in[5];
    const float* el1w = (const float*)d_in[6];
    const float* el1b = (const float*)d_in[7];
    auto Wl = [&](int base, int l) { return (const float*)d_in[base + 2 * l]; };
    auto Bl = [&](int base, int l) { return (const float*)d_in[base + 2 * l + 1]; };
    // v_lins1:8, v_lins2:14, v_lins3:20, v_lins4:26, e_lins:32
    // v_bn_g:38+l, v_bn_b:41+l, e_bn_g:44+l, e_bn_b:47+l

    // --- workspace carve-up ---
    char* base = (char*)d_ws;
    size_t off = 0;
    auto carve = [&](size_t bytes) {
        char* p = base + off;
        off = (off + bytes + 255) & ~(size_t)255;
        return p;
    };
    float* wcur = (float*)carve((size_t)ET * UU * 4);      // 163 MB
    float* tbuf = (float*)carve((size_t)ET * UU * 4);      // 163 MB
    float* xv   = (float*)carve((size_t)NODES * UU * 4);
    float* x1   = (float*)carve((size_t)NODES * UU * 4);
    float* x2   = (float*)carve((size_t)NODES * UU * 4);
    float* x3   = (float*)carve((size_t)NODES * UU * 4);
    float* x4   = (float*)carve((size_t)NODES * UU * 4);
    float* agg  = (float*)carve((size_t)NODES * UU * 4);
    unsigned short* wbf = (unsigned short*)carve((size_t)15 * UU * UU * 2);
    int*   srcA = (int*)carve((size_t)ET * 4);
    int*   dstA = (int*)carve((size_t)ET * 4);
    float* estats = (float*)carve(256 * 4);                // esum | esumsq
    float* esum   = estats, *esumsq = estats + UU;
    float* emu    = (float*)carve(UU * 4);
    float* ersig  = (float*)carve(UU * 4);
    float* nmu    = (float*)carve(UU * 4);
    float* nrsig  = (float*)carve(UU * 4);

    // --- prep: bf16 weights + edge index lists + embeddings ---
    W15 wp;
    for (int l = 0; l < DEPTH; ++l) {
        wp.p[l * 5 + 0] = Wl(8,  l);
        wp.p[l * 5 + 1] = Wl(14, l);
        wp.p[l * 5 + 2] = Wl(20, l);
        wp.p[l * 5 + 3] = Wl(26, l);
        wp.p[l * 5 + 4] = Wl(32, l);
    }
    k_prep_weights<<<(15 * UU * UU) / 256, 256, 0, stream>>>(wp, wbf);
    k_build_edges<<<(ET + 255) / 256, 256, 0, stream>>>(srcA, dstA);
    k_node_embed<<<(NODES * UU) / 256, 256, 0, stream>>>(x, vl0w, vl0b, xv);
    k_edge_embed<<<(int)(((long long)ET * UU) / 256), 256, 0, stream>>>(adj, el0w, el0b, wcur);

    // --- layers ---
    for (int l = 0; l < DEPTH; ++l) {
        k_zero<<<1, 256, 0, stream>>>(estats, 256);

        NodeGemmArgs na;
        na.A = xv;
        for (int j = 0; j < 4; ++j) {
            na.W[j]    = wbf + (size_t)(l * 5 + j) * UU * UU;
            na.out[j]  = (j == 0) ? x1 : (j == 1) ? x2 : (j == 2) ? x3 : x4;
        }
        na.bias[0] = Bl(8, l);  na.bias[1] = Bl(14, l);
        na.bias[2] = Bl(20, l); na.bias[3] = Bl(26, l);
        k_node_gemm<<<dim3(NODES / 16, 4), 256, 0, stream>>>(na);

        k_edge_gemm<<<EGEMM_WGS, 256, 0, stream>>>(
            wcur, wbf + (size_t)(l * 5 + 4) * UU * UU, Bl(32, l),
            x3, x4, srcA, dstA, tbuf, esum, esumsq);

        k_agg<<<NODES, 128, 0, stream>>>(wcur, x2, agg);
        k_node_stats<<<UU, 256, 0, stream>>>(x1, agg, nmu, nrsig);
        k_edge_stats_final<<<1, UU, 0, stream>>>(esum, esumsq, emu, ersig);

        k_node_update<<<(NODES * UU) / 256, 256, 0, stream>>>(
            xv, x1, agg, nmu, nrsig,
            (const float*)d_in[38 + l], (const float*)d_in[41 + l]);
        k_edge_update<<<(int)(((long long)ET * UU) / 256), 256, 0, stream>>>(
            wcur, tbuf, emu, ersig,
            (const float*)d_in[44 + l], (const float*)d_in[47 + l]);
    }

    // --- output projection + diagonal zeros ---
    k_out<<<ET / 8, 256, 0, stream>>>(wcur, el1w, el1b, (float*)d_out);
    k_out_diag<<<(NODES + 255) / 256, 256, 0, stream>>>((float*)d_out);
}
